// MultiHeadGATLayer_27814208209050
// MI455X (gfx1250) — matrix operand order repaired and checked by reading, not compiled, not run on hardware
//
#include <hip/hip_runtime.h>
#include <hip/hip_bf16.h>

typedef __attribute__((ext_vector_type(16))) _Float16 v16h;
typedef __attribute__((ext_vector_type(8)))  float    v8f;

#define NN     4096
#define HEADS  12
#define FOUT   64
#define GAT_ALPHA 0.2f
#define NTILES (NN / 32)

// ---------------- Kernel 0: fold W,a into per-head scalars + x min/max ----------
__global__ __launch_bounds__(256) void gat_scalars(const float* __restrict__ x,
                                                   const float* __restrict__ W,
                                                   const float* __restrict__ a,
                                                   float* __restrict__ ws) {
  __shared__ float smax[256], smin[256];
  const int t = threadIdx.x;
  float mx = -3.0e38f, mn = 3.0e38f;
  for (int i = t; i < NN; i += 256) {
    float v = x[i];
    mx = fmaxf(mx, v); mn = fminf(mn, v);
  }
  smax[t] = mx; smin[t] = mn;
  __syncthreads();
  for (int s = 128; s > 0; s >>= 1) {
    if (t < s) { smax[t] = fmaxf(smax[t], smax[t+s]); smin[t] = fminf(smin[t], smin[t+s]); }
    __syncthreads();
  }
  if (t < 24) {
    int h = t % 12, part = t / 12;            // part 0 -> a_src, 1 -> a_dst
    float acc = 0.f;
    for (int f = 0; f < FOUT; ++f)
      acc = fmaf(W[h*FOUT + f], a[h*2*FOUT + part*FOUT + f], acc);
    ws[part*12 + h] = acc;                    // ws[0..11]=s_h, ws[12..23]=d_h
  }
  if (t == 0) { ws[24] = smax[0]; ws[25] = smin[0]; }
}

// ---------------- Kernel 1: masked softmax row-reduction via WMMA ---------------
// grid = (256 row-tiles, 3 head-groups); block = 128 threads (4 waves).
// Wave w handles head h = 4*blockIdx.y + w for a 16-row tile. The j-reduction
// (num = sum P*x_j, den = sum P) runs through v_wmma_f32_16x16x32_f16 with
// B = [x_j | 1 | 0...]. adj tile staging is software-pipelined through regs.
__global__ __launch_bounds__(128) void gat_main(const float* __restrict__ x,
                                                const int*   __restrict__ adj,
                                                const float* __restrict__ W,
                                                const float* __restrict__ ws,
                                                float*       __restrict__ out) {
  __shared__ float x_lds[NN];          // entire x vector, staged once (16 KB)
  __shared__ int4  adj_s4[128];        // 16 rows x 32 cols int32 (2 KB)
  __shared__ float red[4 * 256];       // per-wave 16x16 accumulator spill
  __shared__ float cbuf[4 * 16];       // per-wave c[row] = num/den
  const int* adj_si = (const int*)adj_s4;

  const int tid   = threadIdx.x;
  const int wave  = tid >> 5;
  const int lane  = tid & 31;
  const int h     = blockIdx.y * 4 + wave;
  const int ibase = blockIdx.x * 16;

  // stage all of x into LDS (vectorized, 8 float4 per thread)
  {
    float4*       xd = (float4*)x_lds;
    const float4* xg = (const float4*)x;
#pragma unroll
    for (int k = 0; k < NN / 4 / 128; ++k)
      xd[tid + k * 128] = xg[tid + k * 128];
  }

  const float sh   = ws[h];
  const float dh   = ws[12 + h];
  const float xmax = ws[24], xmin = ws[25];

  const int   row = lane & 15;
  const float xi  = x[ibase + row];
  const float xis = xi * sh;
  // analytic upper bound of e over j -> one-pass numerically stable softmax
  const float marg   = fmaf(dh, (dh >= 0.f ? xmax : xmin), xis);
  const float mshift = marg > 0.f ? marg : GAT_ALPHA * marg;

  const int   klo  = (lane < 16) ? 0 : 8;      // A-frag K interleave base
  const int   kb   = (lane < 16) ? 0 : 16;     // K half base (unused; B uses klo)
  const int   bn   = lane & 15;                // B column / D column index
  const float sel0 = (bn == 0) ? 1.0f : 0.0f;  // branch-free column select
  const float sel1 = (bn == 1) ? 1.0f : 0.0f;

  v8f acc = {};

  const int sr = tid >> 3, scg = tid & 7;      // staging coords: row, int4 group
  const int* gtile = adj + (size_t)(ibase + sr) * NN + scg * 4;

  int4 cur = *(const int4*)gtile;              // pre-load tile 0
  __syncthreads();                             // x_lds visible

  for (int jt = 0; jt < NTILES; ++jt) {
    const int jbase = jt * 32;
    adj_s4[tid] = cur;                         // commit staged tile
    if (jt + 1 < NTILES)
      cur = *(const int4*)(gtile + (jt + 1) * 32);   // pipeline next tile
    if (jt + 2 < NTILES)
      __builtin_prefetch(gtile + (jt + 2) * 32, 0, 1);
    __syncthreads();                           // adj tile visible

    // B fragment: col0 = x_j, col1 = 1, cols 2..15 = 0  (branch-free)
    v16h bfrag;
#pragma unroll
    for (int e = 0; e < 16; ++e)
      bfrag[e] = (_Float16)fmaf(x_lds[jbase + ((e < 8) ? (klo + e) : (16 + klo + (e - 8)))], sel0, sel1);

    // A fragment: P[i,j] = mask * exp(leakyrelu(xi*s + xj*d) - shift) in (0,1]
    v16h afrag;
#pragma unroll
    for (int e = 0; e < 16; ++e) {
      const int K  = (e < 8) ? (klo + e) : (16 + klo + (e - 8));
      float t  = fmaf(x_lds[jbase + K], dh, xis);
      float lr = t > 0.f ? t : GAT_ALPHA * t;
      float p  = (adj_si[row * 32 + K] > 0) ? __expf(lr - mshift) : 0.f;
      afrag[e] = (_Float16)p;
    }
    acc = __builtin_amdgcn_wmma_f32_16x16x32_f16(false, afrag, false, bfrag,
                                                 (short)0, acc, false, false);
    __syncthreads();                           // done reading tile before overwrite
  }

  // spill 16x16 f32 accumulator (cols 0,1 carry num/den)
#pragma unroll
  for (int r = 0; r < 8; ++r) {
    const int M = (lane < 16) ? r : (8 + r);
    red[wave * 256 + M * 16 + bn] = acc[r];
  }
  __syncthreads();
  if (lane < 16) {
    const float num = red[wave * 256 + lane * 16 + 0];
    const float den = red[wave * 256 + lane * 16 + 1];
    cbuf[wave * 16 + lane] = num / den;
  }
  __syncthreads();

  // out[i, h*64 + f] = c[h,i] * W[h,f]  (coalesced across lanes)
  const float* Wh = W + h * FOUT;
#pragma unroll 4
  for (int t = 0; t < 32; ++t) {
    const int idx = t * 32 + lane;
    const int r = idx >> 6, c = idx & 63;
    out[(size_t)(ibase + r) * (HEADS * FOUT) + h * FOUT + c] =
        cbuf[wave * 16 + r] * Wh[c];
  }
}

extern "C" void kernel_launch(void* const* d_in, const int* in_sizes, int n_in,
                              void* d_out, int out_size, void* d_ws, size_t ws_size,
                              hipStream_t stream) {
  const float* x   = (const float*)d_in[0];
  const int*   adj = (const int*)  d_in[1];
  const float* W   = (const float*)d_in[2];
  const float* a   = (const float*)d_in[3];
  float*       out = (float*)d_out;
  float*       ws  = (float*)d_ws;          // 26 floats used

  gat_scalars<<<1, 256, 0, stream>>>(x, W, a, ws);
  gat_main<<<dim3(NN / 16, 3), 128, 0, stream>>>(x, adj, W, ws, out);
}
